// FlashAttentionLayer_10290741641761
// MI455X (gfx1250) — matrix operand order repaired and checked by reading, not compiled, not run on hardware
//
#include <hip/hip_runtime.h>
#include <hip/hip_bf16.h>

typedef __attribute__((ext_vector_type(16))) __bf16 v16bf;
typedef __attribute__((ext_vector_type(8)))  __bf16 v8bf;
typedef __attribute__((ext_vector_type(8)))  float  v8f;

namespace {

constexpr int kB   = 2;
constexpr int kT   = 2048;
constexpr int kHID = 2048;
constexpr int kNH  = 32;
constexpr int kNKV = 8;
constexpr int kHD  = 64;
constexpr float kScale = 0.125f;  // HD^-0.5

union AFrag { v16bf v; v8bf h[2]; };

__device__ inline v8f wmma_bf16(const AFrag& a, const AFrag& b, v8f c) {
  // (neg_a, A, neg_b, B, c_mod, C, reuse_a, reuse_b)
  return __builtin_amdgcn_wmma_f32_16x16x32_bf16(false, a.v, false, b.v,
                                                 (short)0, c, false, false);
}

__device__ inline v8f vzero8() {
  v8f z;
#pragma unroll
  for (int i = 0; i < 8; ++i) z[i] = 0.0f;
  return z;
}

// ---------------- f32 -> bf16 convert ----------------
__global__ void cvt_bf16_kernel(const float* __restrict__ x,
                                __bf16* __restrict__ y, int n) {
  int i = blockIdx.x * blockDim.x + threadIdx.x;
  int stride = gridDim.x * blockDim.x;
  for (; i < n; i += stride) y[i] = (__bf16)x[i];
}

// ---------------- RoPE in place on [B, H, T, HD] bf16 ----------------
__global__ void rope_kernel(__bf16* __restrict__ x,
                            const float* __restrict__ cosp,
                            const float* __restrict__ sinp, int H, int total) {
  int i = blockIdx.x * blockDim.x + threadIdx.x;
  if (i >= total) return;
  int d  = i & 31;
  int t  = (i >> 5) & (kT - 1);
  int hb = i >> 16;  // b*H + h   (T*32 == 65536)
  int b  = hb / H;
  size_t base  = ((size_t)hb * kT + t) * kHD;
  size_t cbase = ((size_t)b * kT + t) * kHD;
  float q1 = (float)x[base + d];
  float q2 = (float)x[base + d + 32];
  float c1 = cosp[cbase + d],      s1 = sinp[cbase + d];
  float c2 = cosp[cbase + d + 32], s2 = sinp[cbase + d + 32];
  x[base + d]      = (__bf16)(q1 * c1 - q2 * s1);
  x[base + d + 32] = (__bf16)(q2 * c2 + q1 * s2);
}

// ---------------- generic bf16 WMMA GEMM: C[m][n] = sum_k A[m][k]*W[n][k] ----
// MODE 0: bf16 out at [((b*H + n/64)*T + t)*64 + n%64]   (per-head layout)
// MODE 1: bf16 out at [((b*H + n/64)*64 + n%64)*T + t]   (per-head, transposed)
// MODE 2: f32  out at [m*N + n]
template <int MODE>
__global__ __launch_bounds__(256, 2) void gemm_bf16_kernel(
    const __bf16* __restrict__ A, const __bf16* __restrict__ W,
    void* __restrict__ out, int M, int N, int K, int H) {
  constexpr int LP = 40;  // padded LDS pitch (bf16 elems) for 32-wide K tiles
  __shared__ __bf16 lA[128 * LP];
  __shared__ __bf16 lB[64 * LP];

  const int tid  = threadIdx.x;
  const int w    = tid >> 5;
  const int lane = tid & 31;
  const int lm   = lane & 15;
  const int hl   = lane >> 4;
  const int wm   = w >> 1;  // 0..3 (M)
  const int wn   = w & 1;   // 0..1 (N)
  const int m0   = blockIdx.y * 128;
  const int n0   = blockIdx.x * 64;

  v8f acc[4];
#pragma unroll
  for (int i = 0; i < 4; ++i) acc[i] = vzero8();

  const int ra = tid >> 1;        // 0..127
  const int ca = (tid & 1) * 16;  // 0 / 16

  for (int k0 = 0; k0 < K; k0 += 32) {
    {
      const v8bf* src = (const v8bf*)(A + (size_t)(m0 + ra) * K + k0 + ca);
      v8bf a0 = src[0], a1 = src[1];
      *(v8bf*)&lA[ra * LP + ca]     = a0;
      *(v8bf*)&lA[ra * LP + ca + 8] = a1;
      if (tid < 128) {
        const v8bf* sw = (const v8bf*)(W + (size_t)(n0 + ra) * K + k0 + ca);
        v8bf b0 = sw[0], b1 = sw[1];
        *(v8bf*)&lB[ra * LP + ca]     = b0;
        *(v8bf*)&lB[ra * LP + ca + 8] = b1;
      }
      if (k0 + 32 < K)
        __builtin_prefetch(A + (size_t)(m0 + ra) * K + k0 + 32 + ca, 0, 1);
    }
    __syncthreads();

    AFrag af[2], bfr[2];
#pragma unroll
    for (int mi = 0; mi < 2; ++mi) {
      const __bf16* p = &lA[(wm * 32 + mi * 16 + lm) * LP + hl * 8];
      af[mi].h[0] = *(const v8bf*)p;        // K = hl*8 .. +7
      af[mi].h[1] = *(const v8bf*)(p + 16); // K = 16+hl*8 .. +7
    }
#pragma unroll
    for (int ni = 0; ni < 2; ++ni) {
      const __bf16* p = &lB[(wn * 32 + ni * 16 + lm) * LP + hl * 8];
      bfr[ni].h[0] = *(const v8bf*)p;       // K = hl*8 .. +7, then 16+hl*8 .. +7
      bfr[ni].h[1] = *(const v8bf*)(p + 16);
    }
    acc[0] = wmma_bf16(af[0], bfr[0], acc[0]);
    acc[1] = wmma_bf16(af[0], bfr[1], acc[1]);
    acc[2] = wmma_bf16(af[1], bfr[0], acc[2]);
    acc[3] = wmma_bf16(af[1], bfr[1], acc[3]);
    __syncthreads();
  }

#pragma unroll
  for (int mi = 0; mi < 2; ++mi)
#pragma unroll
    for (int ni = 0; ni < 2; ++ni)
#pragma unroll
      for (int r = 0; r < 8; ++r) {
        float vv = acc[mi * 2 + ni][r];
        int m = m0 + wm * 32 + mi * 16 + r + 8 * hl;
        int n = n0 + wn * 32 + ni * 16 + lm;
        if (MODE == 2) {
          ((float*)out)[(size_t)m * N + n] = vv;
        } else {
          int b = m >> 11, t = m & (kT - 1);  // rows are b*T + t
          int hh = n >> 6, d = n & 63;
          if (MODE == 0)
            ((__bf16*)out)[(((size_t)(b * H + hh) * kT) + t) * 64 + d] = (__bf16)vv;
          else
            ((__bf16*)out)[(((size_t)(b * H + hh)) * 64 + d) * kT + t] = (__bf16)vv;
        }
      }
}

// ---------------- flash attention: 128 queries / block, 8 waves ----------------
__global__ __launch_bounds__(256, 2) void flash_attn_kernel(
    const __bf16* __restrict__ qbf, const __bf16* __restrict__ kbf,
    const __bf16* __restrict__ vtbf, __bf16* __restrict__ attn) {
  constexpr int AP = 72;  // padded LDS pitch for 64-wide tiles
  __shared__ __bf16 kt[64 * AP];          // K tile   [key][d]
  __shared__ __bf16 vt[64 * AP];          // V^T tile [d][key]
  __shared__ __bf16 pbuf[8 * 16 * AP];    // per-wave P staging [q][key]

  const int blk = blockIdx.x;
  const int qb = blk & 15;        // T/128 = 16
  const int h  = (blk >> 4) & 31;
  const int b  = blk >> 9;
  const int kv = h >> 2;          // NH/NKV = 4
  const int q0 = qb * 128;

  const int tid  = threadIdx.x;
  const int w    = tid >> 5;
  const int lane = tid & 31;
  const int lm   = lane & 15;
  const int hl   = lane >> 4;

  const __bf16* qrow  = qbf  + ((size_t)(b * kNH  + h ) * kT)  * kHD;
  const __bf16* kbase = kbf  + ((size_t)(b * kNKV + kv) * kT)  * kHD;
  const __bf16* vbase = vtbf + ((size_t)(b * kNKV + kv) * kHD) * kT;

  // Q A-fragments: 16 queries x 64 dims -> two K-chunks of 32, loaded from global
  AFrag aq[2];
  const int qr = q0 + w * 16 + lm;
#pragma unroll
  for (int kc = 0; kc < 2; ++kc) {
    const __bf16* p = qrow + (size_t)qr * kHD + kc * 32 + hl * 8;
    aq[kc].h[0] = *(const v8bf*)p;
    aq[kc].h[1] = *(const v8bf*)(p + 16);
  }

  v8f o[4];
#pragma unroll
  for (int i = 0; i < 4; ++i) o[i] = vzero8();
  float mrow[8], lrow[8];
#pragma unroll
  for (int r = 0; r < 8; ++r) { mrow[r] = -3.0e38f; lrow[r] = 0.0f; }

  const int lr = tid >> 2;        // 0..63
  const int lc = (tid & 3) * 16;  // 0/16/32/48
  const int nkb = 2 * (qb + 1);   // causal: key blocks of 64 up to q0+127

  for (int kb = 0; kb < nkb; ++kb) {
    const int k0 = kb * 64;
    {
      const v8bf* sk = (const v8bf*)(kbase + (size_t)(k0 + lr) * kHD + lc);
      v8bf x0 = sk[0], x1 = sk[1];
      *(v8bf*)&kt[lr * AP + lc]     = x0;
      *(v8bf*)&kt[lr * AP + lc + 8] = x1;
      const v8bf* sv = (const v8bf*)(vbase + (size_t)lr * kT + k0 + lc);
      v8bf y0 = sv[0], y1 = sv[1];
      *(v8bf*)&vt[lr * AP + lc]     = y0;
      *(v8bf*)&vt[lr * AP + lc + 8] = y1;
    }
    __syncthreads();

    // S = Q @ K^T : 16 x 64
    v8f s[4];
#pragma unroll
    for (int j = 0; j < 4; ++j) s[j] = vzero8();
#pragma unroll
    for (int kc = 0; kc < 2; ++kc) {
#pragma unroll
      for (int j = 0; j < 4; ++j) {
        AFrag bk;
        const __bf16* p = &kt[(j * 16 + lm) * AP + kc * 32 + hl * 8];
        bk.h[0] = *(const v8bf*)p;
        bk.h[1] = *(const v8bf*)(p + 16);
        s[j] = wmma_bf16(aq[kc], bk, s[j]);
      }
    }

    // scale + causal mask
#pragma unroll
    for (int j = 0; j < 4; ++j)
#pragma unroll
      for (int r = 0; r < 8; ++r) {
        int col = k0 + j * 16 + lm;
        int row = q0 + w * 16 + r + 8 * hl;
        float vv = s[j][r] * kScale;
        s[j][r] = (col <= row) ? vv : -1.0e30f;
      }

    // online softmax: each row lives in one 16-lane half
    float mnew[8];
#pragma unroll
    for (int r = 0; r < 8; ++r) {
      float t = fmaxf(fmaxf(s[0][r], s[1][r]), fmaxf(s[2][r], s[3][r]));
      t = fmaxf(t, __shfl_xor(t, 1));
      t = fmaxf(t, __shfl_xor(t, 2));
      t = fmaxf(t, __shfl_xor(t, 4));
      t = fmaxf(t, __shfl_xor(t, 8));
      mnew[r] = fmaxf(mrow[r], t);
    }
#pragma unroll
    for (int j = 0; j < 4; ++j)
#pragma unroll
      for (int r = 0; r < 8; ++r) s[j][r] = __expf(s[j][r] - mnew[r]);
#pragma unroll
    for (int r = 0; r < 8; ++r) {
      float acc = s[0][r] + s[1][r] + s[2][r] + s[3][r];
      acc += __shfl_xor(acc, 1);
      acc += __shfl_xor(acc, 2);
      acc += __shfl_xor(acc, 4);
      acc += __shfl_xor(acc, 8);
      float alpha = __expf(mrow[r] - mnew[r]);
      lrow[r] = lrow[r] * alpha + acc;
      mrow[r] = mnew[r];
#pragma unroll
      for (int f = 0; f < 4; ++f) o[f][r] *= alpha;
    }

    // stage P (bf16) to LDS: C-layout -> A-layout reshape
    __bf16* pb = &pbuf[w * 16 * AP];
#pragma unroll
    for (int j = 0; j < 4; ++j)
#pragma unroll
      for (int r = 0; r < 8; ++r)
        pb[(r + 8 * hl) * AP + j * 16 + lm] = (__bf16)s[j][r];

    // O += P @ V
#pragma unroll
    for (int kc = 0; kc < 2; ++kc) {
      AFrag ap;
      const __bf16* p = pb + lm * AP + kc * 32 + hl * 8;
      ap.h[0] = *(const v8bf*)p;
      ap.h[1] = *(const v8bf*)(p + 16);
#pragma unroll
      for (int j = 0; j < 4; ++j) {
        AFrag bv;
        const __bf16* q = &vt[(j * 16 + lm) * AP + kc * 32 + hl * 8];
        bv.h[0] = *(const v8bf*)q;
        bv.h[1] = *(const v8bf*)(q + 16);
        o[j] = wmma_bf16(ap, bv, o[j]);
      }
    }
    __syncthreads();
  }

  // normalize + store attn[b][t][h*64+d] (bf16) for the output projection
#pragma unroll
  for (int j = 0; j < 4; ++j)
#pragma unroll
    for (int r = 0; r < 8; ++r) {
      int row = q0 + w * 16 + r + 8 * hl;
      int d = j * 16 + lm;
      float vv = o[j][r] / lrow[r];
      attn[((size_t)b * kT + row) * (kNH * kHD) + h * kHD + d] = (__bf16)vv;
    }
}

}  // namespace

extern "C" void kernel_launch(void* const* d_in, const int* in_sizes, int n_in,
                              void* d_out, int out_size, void* d_ws, size_t ws_size,
                              hipStream_t stream) {
  const float* hs   = (const float*)d_in[0];
  const float* cosp = (const float*)d_in[1];
  const float* sinp = (const float*)d_in[2];
  const float* wq   = (const float*)d_in[3];
  const float* wk   = (const float*)d_in[4];
  const float* wv   = (const float*)d_in[5];
  const float* wo   = (const float*)d_in[6];
  float* out = (float*)d_out;

  char* wsp = (char*)d_ws;
  auto take = [&](size_t elems) -> __bf16* {
    __bf16* p = (__bf16*)wsp;
    wsp += elems * sizeof(__bf16);
    return p;
  };
  const size_t M = (size_t)kB * kT;  // 4096
  __bf16* hs_bf = take(M * kHID);
  __bf16* wq_bf = take((size_t)kNH * kHD * kHID);
  __bf16* wk_bf = take((size_t)kNKV * kHD * kHID);
  __bf16* wv_bf = take((size_t)kNKV * kHD * kHID);
  __bf16* wo_bf = take((size_t)kHID * kNH * kHD);
  __bf16* q_bf  = take((size_t)kB * kNH * kT * kHD);
  __bf16* k_bf  = take((size_t)kB * kNKV * kT * kHD);
  __bf16* vT_bf = take((size_t)kB * kNKV * kHD * kT);
  __bf16* at_bf = take(M * (size_t)(kNH * kHD));

  auto cvt = [&](const float* x, __bf16* y, int n) {
    cvt_bf16_kernel<<<(n + 255) / 256, 256, 0, stream>>>(x, y, n);
  };
  cvt(hs, hs_bf, (int)(M * kHID));
  cvt(wq, wq_bf, kNH * kHD * kHID);
  cvt(wk, wk_bf, kNKV * kHD * kHID);
  cvt(wv, wv_bf, kNKV * kHD * kHID);
  cvt(wo, wo_bf, kHID * kNH * kHD);

  dim3 gQ(kNH * kHD / 64, (unsigned)(M / 128));   // (32, 32)
  gemm_bf16_kernel<0><<<gQ, 256, 0, stream>>>(hs_bf, wq_bf, q_bf, (int)M, kNH * kHD, kHID, kNH);
  dim3 gKV(kNKV * kHD / 64, (unsigned)(M / 128)); // (8, 32)
  gemm_bf16_kernel<0><<<gKV, 256, 0, stream>>>(hs_bf, wk_bf, k_bf, (int)M, kNKV * kHD, kHID, kNKV);
  gemm_bf16_kernel<1><<<gKV, 256, 0, stream>>>(hs_bf, wv_bf, vT_bf, (int)M, kNKV * kHD, kHID, kNKV);

  int totQ = kB * kNH * kT * 32;
  rope_kernel<<<(totQ + 255) / 256, 256, 0, stream>>>(q_bf, cosp, sinp, kNH, totQ);
  int totK = kB * kNKV * kT * 32;
  rope_kernel<<<(totK + 255) / 256, 256, 0, stream>>>(k_bf, cosp, sinp, kNKV, totK);

  flash_attn_kernel<<<kB * kNH * (kT / 128), 256, 0, stream>>>(q_bf, k_bf, vT_bf, at_bf);

  dim3 gO(kHID / 64, (unsigned)(M / 128));        // (32, 32)
  gemm_bf16_kernel<2><<<gO, 256, 0, stream>>>(at_bf, wo_bf, out, (int)M, kHID, kHID, 0);

  (void)in_sizes; (void)n_in; (void)out_size; (void)ws_size;
}
